// DeltaGN_60498909331860
// MI455X (gfx1250) — matrix operand order repaired and checked by reading, not compiled, not run on hardware
//
#include <hip/hip_runtime.h>
#include <hip/hip_bf16.h>

typedef __attribute__((ext_vector_type(16))) _Float16 v16h;
typedef __attribute__((ext_vector_type(8)))  _Float16 v8h;
typedef __attribute__((ext_vector_type(8)))  float    v8f;

#define BB   2
#define NN   4096
#define EE   262144
#define ED   150
#define EDP  160      // ED padded to 10 x 16
#define HALFBOX 3.0f
#define BOX     6.0f

// ---- LDS layout (bytes) -------------------------------------------------
// floats: agg[2][160] | Nin[2][160] | nhA[2][100] | nhB[2][100]
// halfs : wsw[60][32][16]  (B-fragment-swizzled weights: tiles 0..9 = eW1
//                           (kc=0,j), tiles 10..59 = eW2 (kc*10+j))
//         ast[8][5][32][16] (A-fragment-swizzled per-wave activations)
#define OFF_AGG  0
#define OFF_NIN  1280
#define OFF_NHA  2560
#define OFF_NHB  3360
#define OFF_WSW  4160      // 60*32*16*2 = 61440 B   (32B aligned)
#define OFF_AST  65600     // 8*5*32*16*2 = 40960 B  (32B aligned)
#define SMEM_BYTES 106560

// Fragment-swizzled load: each lane's 16 halves are contiguous & 32B aligned,
// so this lowers to 2x ds_load_b128.
__device__ __forceinline__ v16h ld_frag(const _Float16* base, int tile, int lane) {
  return *(const v16h*)(base + (size_t)(tile * 32 + lane) * 16);
}

__global__ __launch_bounds__(256)
void deltagn_fused(const float* __restrict__ V,   const float* __restrict__ dt,
                   const float* __restrict__ eW1, const float* __restrict__ eb1,
                   const float* __restrict__ eW2, const float* __restrict__ eb2,
                   const float* __restrict__ nW1, const float* __restrict__ nb1,
                   const float* __restrict__ nW2, const float* __restrict__ nb2,
                   const float* __restrict__ nW3, const float* __restrict__ nb3,
                   const float* __restrict__ lW,  const float* __restrict__ lb,
                   const int* __restrict__ R_s,   const int* __restrict__ R_r,
                   float* __restrict__ out)
{
  extern __shared__ char smem[];
  float*     agg = (float*)(smem + OFF_AGG);      // [2][160]
  float*     Nin = (float*)(smem + OFF_NIN);      // [2][160] (154 used)
  float*     nhA = (float*)(smem + OFF_NHA);      // [2][100]
  float*     nhB = (float*)(smem + OFF_NHB);      // [2][100]
  _Float16*  wsw = (_Float16*)(smem + OFF_WSW);   // [60][32][16]
  _Float16*  ast = (_Float16*)(smem + OFF_AST);   // [8][5][32][16]

  const int tid  = threadIdx.x;
  const int wave = tid >> 5;
  const int lane = tid & 31;

  const long long geBase = (long long)blockIdx.x * 128;   // first edge (flat b*E+e)
  const int b     = (int)(geBase / EE);
  const int eInB  = (int)(geBase % EE);
  const int node0 = eInB >> 6;                            // first of the 2 nodes
  const float dtb = dt[b];

  // ---- stage weights directly in B-fragment order -----------------------
  // B-frag semantics (chunk kOff, col tile nOff): lane (n=lane&15, kh=lane>>4),
// element i  <->  B[kOff + kh*8 + (i&7) + (i>>3)*16][nOff + n]
  for (int idx = tid; idx < 60 * 32 * 16; idx += 256) {
    const int t    = idx >> 9;          // /512
    const int ln   = (idx >> 4) & 31;
    const int ii   = idx & 15;
    const int n    = ((t < 10 ? t : (t - 10) % 10) << 4) + (ln & 15);
    const int k    = (t < 10 ? 0 : ((t - 10) / 10) * 32) + ((ln >> 4) << 3) + (ii & 7) + ((ii >> 3) << 4);
    float v = 0.0f;
    if (t < 10) { if (k < 9  && n < ED) v = eW1[k * ED + n]; }
    else        { if (k < ED && n < ED) v = eW2[k * ED + n]; }
    wsw[idx] = (_Float16)v;
  }
  for (int i = tid; i < 2 * EDP; i += 256) agg[i] = 0.0f;

  _Float16* aw = ast + (size_t)wave * 5 * 32 * 16;   // this wave's A stage

  // ---- build Ein tile directly in A-fragment order (chunk kc=0) ----------
  // A-frag semantics: lane' (m=lane'&15, hi=lane'>>4), element e <-> K where
  // e<8: K=(hi?8:0)+e ; e>=8: K=(hi?24:16)+(e-8)
  if (lane < 16) {
    const long long ge = geBase + wave * 16 + lane;
    const int s = R_s[ge];
    const int r = R_r[ge];
    const float* Vs = V + ((size_t)b * NN + s) * 5;
    const float* Vr = V + ((size_t)b * NN + r) * 5;
    float ein[9];
    ein[0] = Vs[0]; ein[1] = Vs[3]; ein[2] = Vs[4];
    ein[3] = Vr[0]; ein[4] = Vr[3]; ein[5] = Vr[4];
    float d0 = Vs[1] - Vr[1];
    d0 = (d0 >  HALFBOX) ? d0 - BOX : d0;
    d0 = (d0 <= -HALFBOX) ? d0 + BOX : d0;
    float d1 = Vs[2] - Vr[2];
    d1 = (d1 >  HALFBOX) ? d1 - BOX : d1;
    d1 = (d1 <= -HALFBOX) ? d1 + BOX : d1;
    ein[6] = d0; ein[7] = d1; ein[8] = dtb;

    v8h z8 = v8h{};
    v8h lo;                       // K=0..7
#pragma unroll
    for (int i = 0; i < 8; ++i) lo[i] = (_Float16)ein[i];
    v8h k8 = z8; k8[0] = (_Float16)ein[8];   // K=8..15 -> only K=8 nonzero
    _Float16* r0 = aw + (size_t)lane * 16;          // lane'=m   (hi=0)
    _Float16* r1 = aw + (size_t)(lane + 16) * 16;   // lane'=m+16(hi=1)
    *(v8h*)(r0)     = lo;   // e0..7  : K 0..7
    *(v8h*)(r0 + 8) = z8;   // e8..15 : K 16..23
    *(v8h*)(r1)     = k8;   // e0..7  : K 8..15
    *(v8h*)(r1 + 8) = z8;   // e8..15 : K 24..31
  }
  __syncthreads();

  const int colB = lane & 15;
  const int rOff = (lane >> 4) ? 8 : 0;

  // ---- edge layer 1: (16x32) @ (32x160) ---------------------------------
  {
    v16h a = ld_frag(aw, 0, lane);   // kc=0 chunk of A stage
    v8f c1[10];
#pragma unroll
    for (int j = 0; j < 10; ++j) c1[j] = v8f{};
#pragma unroll
    for (int j = 0; j < 10; ++j) {
      v16h bf = ld_frag(wsw, j, lane);
      c1[j] = __builtin_amdgcn_wmma_f32_16x16x32_f16(false, a, false, bf,
                                                     (short)0, c1[j], false, false);
    }
    // bias + relu, store h back in A-fragment order (overwrites Ein chunk 0)
#pragma unroll
    for (int j = 0; j < 10; ++j) {
      const int col = 16 * j + colB;
      const float bia = (col < ED) ? eb1[col] : 0.0f;
      const int kc  = j >> 1;
      const int kk  = ((j & 1) << 4) + colB;       // K within chunk
      const int grp = kk >> 3;
      const int hi  = grp & 1;
      const int e   = (kk & 7) + ((grp >> 1) << 3);
#pragma unroll
      for (int r = 0; r < 8; ++r) {
        float v = c1[j][r] + bia;
        v = v > 0.0f ? v : 0.0f;
        const int row = rOff + r;                  // M of this C element
        aw[(size_t)(kc * 32 + row + (hi << 4)) * 16 + e] = (_Float16)v;
      }
    }
  }
  __syncthreads();

  // ---- edge layer 2: (16x160) @ (160x160), 5 K-chunks --------------------
  {
    v8f c2[10];
#pragma unroll
    for (int j = 0; j < 10; ++j) c2[j] = v8f{};
#pragma unroll
    for (int kc = 0; kc < 5; ++kc) {
      v16h a2 = ld_frag(aw, kc, lane);
#pragma unroll
      for (int j = 0; j < 10; ++j) {
        v16h bf = ld_frag(wsw, 10 + kc * 10 + j, lane);
        c2[j] = __builtin_amdgcn_wmma_f32_16x16x32_f16(false, a2, false, bf,
                                                       (short)0, c2[j], false, false);
      }
    }
    // bias + relu + row-sum (reduce over the 16 edges of this tile) -> agg
    const int nodeLocal = wave >> 2;   // waves 0-3 -> node 0, 4-7 -> node 1
#pragma unroll
    for (int j = 0; j < 10; ++j) {
      const int col = 16 * j + colB;
      if (col < ED) {
        const float bia = eb2[col];
        float sum = 0.0f;
#pragma unroll
        for (int r = 0; r < 8; ++r) {
          float v = c2[j][r] + bia;
          sum += v > 0.0f ? v : 0.0f;
        }
        atomicAdd(&agg[nodeLocal * EDP + col], sum);   // ds_add_f32
      }
    }
  }
  __syncthreads();

  // ---- node stage: 2 nodes, tiny MLP 154->100->100->100->4 ---------------
  for (int i = tid; i < 2 * 154; i += 256) {
    const int nd = i / 154, k = i % 154;
    const float* Vrow = V + ((size_t)b * NN + node0 + nd) * 5;
    float v;
    if      (k == 0)  v = Vrow[0];
    else if (k == 1)  v = Vrow[3];
    else if (k == 2)  v = Vrow[4];
    else if (k < 153) v = agg[nd * EDP + (k - 3)];
    else              v = dtb;
    Nin[nd * EDP + k] = v;
  }
  __syncthreads();

  for (int i = tid; i < 200; i += 256) {
    const int nd = i / 100, u = i % 100;
    float acc = nb1[u];
    for (int k = 0; k < 154; ++k) acc += Nin[nd * EDP + k] * nW1[k * 100 + u];
    nhA[i] = acc > 0.0f ? acc : 0.0f;
  }
  __syncthreads();

  for (int i = tid; i < 200; i += 256) {
    const int nd = i / 100, u = i % 100;
    float acc = nb2[u];
    for (int k = 0; k < 100; ++k) acc += nhA[nd * 100 + k] * nW2[k * 100 + u];
    nhB[i] = acc > 0.0f ? acc : 0.0f;
  }
  __syncthreads();

  for (int i = tid; i < 200; i += 256) {
    const int nd = i / 100, u = i % 100;
    float acc = nb3[u];
    for (int k = 0; k < 100; ++k) acc += nhB[nd * 100 + k] * nW3[k * 100 + u];
    nhA[i] = acc > 0.0f ? acc : 0.0f;
  }
  __syncthreads();

  for (int i = tid; i < 8; i += 256) {
    const int nd = i / 4, j = i % 4;
    float acc = lb[j];
    for (int u = 0; u < 100; ++u) acc += nhA[nd * 100 + u] * lW[u * 4 + j];
    const float* Vrow = V + ((size_t)b * NN + node0 + nd) * 5;
    float nv = Vrow[1 + j] + acc;
    if (j < 2) {
      if (nv >=  HALFBOX) nv -= BOX;
      if (nv <  -HALFBOX) nv += BOX;
    }
    out[((size_t)b * NN + node0 + nd) * 4 + j] = nv;
  }
}

extern "C" void kernel_launch(void* const* d_in, const int* in_sizes, int n_in,
                              void* d_out, int out_size, void* d_ws, size_t ws_size,
                              hipStream_t stream) {
  (void)in_sizes; (void)n_in; (void)out_size; (void)d_ws; (void)ws_size;
  const float* V   = (const float*)d_in[0];
  const float* dt  = (const float*)d_in[1];
  const float* eW1 = (const float*)d_in[2];
  const float* eb1 = (const float*)d_in[3];
  const float* eW2 = (const float*)d_in[4];
  const float* eb2 = (const float*)d_in[5];
  const float* nW1 = (const float*)d_in[6];
  const float* nb1 = (const float*)d_in[7];
  const float* nW2 = (const float*)d_in[8];
  const float* nb2 = (const float*)d_in[9];
  const float* nW3 = (const float*)d_in[10];
  const float* nb3 = (const float*)d_in[11];
  const float* lW  = (const float*)d_in[12];
  const float* lb  = (const float*)d_in[13];
  const int*   R_s = (const int*)d_in[14];
  const int*   R_r = (const int*)d_in[15];
  float* out = (float*)d_out;

  // >64KB dynamic LDS: raise the per-kernel cap (CDNA5 WGP allows 320KB/WG).
  (void)hipFuncSetAttribute((const void*)deltagn_fused,
                            hipFuncAttributeMaxDynamicSharedMemorySize, SMEM_BYTES);

  const int nBlocks = (BB * EE) / 128;   // 4096 blocks, 2 nodes each
  deltagn_fused<<<nBlocks, 256, SMEM_BYTES, stream>>>(
      V, dt, eW1, eb1, eW2, eb2, nW1, nb1, nW2, nb2, nW3, nb3, lW, lb,
      R_s, R_r, out);
}
